// NodeEncoder_82497731822002
// MI455X (gfx1250) — matrix operand order repaired and checked by reading, not compiled, not run on hardware
//
#include <hip/hip_runtime.h>

// ---------------------------------------------------------------------------
// GCN forward on MI455X (gfx1250, wave32), bf16 WMMA with f32 accumulate.
//   s1  = x  @ W1                      (bf16 out)
//   h   = relu(adj @ s1 + b1)          (bf16 out)
//   s2  = h  @ W2                      (bf16 out)
//   out = adj @ s2 + b2                (f32 out)
// Double-buffered LDS pipeline, one barrier per K-step; N/K are template
// constants so all loop addressing strength-reduces to constant increments.
// ---------------------------------------------------------------------------

typedef __attribute__((ext_vector_type(16))) __bf16        v16bf;
typedef __attribute__((ext_vector_type(8)))  __bf16        v8bf;
typedef __attribute__((ext_vector_type(8)))  float         v8f;
typedef __attribute__((ext_vector_type(8)))  unsigned short us8;
typedef __attribute__((ext_vector_type(4)))  unsigned short us4;

__device__ __forceinline__ unsigned short f32_to_bf16_rne(float f) {
  unsigned u = __float_as_uint(f);
  u += 0x7FFFu + ((u >> 16) & 1u);   // round-to-nearest-even
  return (unsigned short)(u >> 16);
}

// ---------------------------------------------------------------------------
// f32 -> bf16 conversion, 4 elements/thread
// ---------------------------------------------------------------------------
__global__ __launch_bounds__(256)
void cvt_f32_bf16(const float* __restrict__ src,
                  unsigned short* __restrict__ dst,
                  long long nvec4) {
  long long i = (long long)blockIdx.x * 256 + threadIdx.x;
  if (i >= nvec4) return;
  float4 v = ((const float4*)src)[i];
  us4 o;
  o[0] = f32_to_bf16_rne(v.x);
  o[1] = f32_to_bf16_rne(v.y);
  o[2] = f32_to_bf16_rne(v.z);
  o[3] = f32_to_bf16_rne(v.w);
  ((us4*)dst)[i] = o;
}

// ---------------------------------------------------------------------------
// Tiled bf16 GEMM: C[M,N] = A[M,K] @ B[K,N] (+bias) (+relu)
// Block tile 128x128, BK=32, double-buffered LDS. 256 threads = 8 waves
// (4 rows x 2 cols); each wave computes 32x64 via 2x4 WMMA accumulators.
// ---------------------------------------------------------------------------
template<int N, int K, bool HAS_BIAS, bool RELU, bool OUT_F32>
__global__ __launch_bounds__(256)
void gcn_gemm_bf16(const unsigned short* __restrict__ A,
                   const unsigned short* __restrict__ B,
                   const float* __restrict__ bias,
                   unsigned short* __restrict__ Cb,
                   float* __restrict__ Cf) {
  __shared__ __align__(32) unsigned short As[2][128][32];   // [buf][m][k]
  __shared__ __align__(32) unsigned short BsT[2][128][32];  // [buf][n][k]

  const int tid  = threadIdx.x;
  const int lane = tid & 31;
  const int wave = tid >> 5;
  const int wr   = wave >> 1;      // wave row 0..3  -> 32 rows each
  const int wc   = wave & 1;       // wave col 0..1  -> 64 cols each
  const int kh   = lane >> 4;      // lane half selects K sub-range
  const int r    = lane & 15;

  const int blockM = blockIdx.y * 128;
  const int blockN = blockIdx.x * 128;

  // A staging: thread -> (row, 16-elem K segment)
  const int a_row = tid >> 1;            // 0..127
  const int a_k   = (tid & 1) * 16;      // 0 or 16
  // B staging: thread -> (even k pair, 8-elem N segment), transposed store
  const int b_kp  = (tid >> 4) << 1;     // 0,2,...,30
  const int b_n   = (tid & 15) * 8;      // 0..120

  v8f acc[2][4];
  const v8f vzero = {0.f,0.f,0.f,0.f,0.f,0.f,0.f,0.f};
#pragma unroll
  for (int mi = 0; mi < 2; ++mi)
#pragma unroll
    for (int ni = 0; ni < 4; ++ni) acc[mi][ni] = vzero;

  // ---- pipeline state: explicitly incremented pointers ------------------
  const us8* aptr = (const us8*)(A + (size_t)(blockM + a_row) * K + a_k);
  const unsigned short* bptr = B + (size_t)b_kp * N + blockN + b_n;

  us8 av0, av1, bv0, bv1;

  auto load_global = [&]() {
    av0 = aptr[0];
    av1 = aptr[1];
    bv0 = *(const us8*)bptr;
    bv1 = *(const us8*)(bptr + N);     // constant immediate offset
    aptr += 4;                         // +32 bf16 along K
    bptr += 32 * N;                    // +32 rows of B
  };

  auto stage_lds = [&](int buf) {
    *(us8*)&As[buf][a_row][a_k]     = av0;
    *(us8*)&As[buf][a_row][a_k + 8] = av1;
#pragma unroll
    for (int j = 0; j < 8; ++j) {
      unsigned p = (unsigned)bv0[j] | ((unsigned)bv1[j] << 16);
      *(unsigned*)&BsT[buf][b_n + j][b_kp] = p;   // [n][k], [n][k+1]
    }
  };

  auto compute = [&](int buf) {
    v16bf afrag[2], bfrag[4];
#pragma unroll
    for (int mi = 0; mi < 2; ++mi) {
      const int row = wr * 32 + mi * 16 + r;
      v8bf lo = *(const v8bf*)&As[buf][row][kh * 8];        // K 0-7 / 8-15
      v8bf hi = *(const v8bf*)&As[buf][row][16 + kh * 8];   // K 16-23 / 24-31
      afrag[mi] = __builtin_shufflevector(lo, hi,
          0,1,2,3,4,5,6,7,8,9,10,11,12,13,14,15);
    }
#pragma unroll
    for (int ni = 0; ni < 4; ++ni) {
      const int n = wc * 64 + ni * 16 + r;
      v8bf lo = *(const v8bf*)&BsT[buf][n][kh * 8];
      v8bf hi = *(const v8bf*)&BsT[buf][n][16 + kh * 8];
      bfrag[ni] = __builtin_shufflevector(lo, hi,
          0,1,2,3,4,5,6,7,8,9,10,11,12,13,14,15);
    }
#pragma unroll
    for (int mi = 0; mi < 2; ++mi)
#pragma unroll
      for (int ni = 0; ni < 4; ++ni)
        acc[mi][ni] = __builtin_amdgcn_wmma_f32_16x16x32_bf16(
            false, afrag[mi], false, bfrag[ni],
            (short)0, acc[mi][ni], false, false);
  };

  // ---- software pipeline: prologue fills buffer 0 -----------------------
  load_global();
  stage_lds(0);
  __syncthreads();

  int cur = 0;
#pragma unroll 1
  for (int k0 = 32; k0 < K; k0 += 32) {
    load_global();         // issue global loads for next tile
    compute(cur);          // WMMAs on current tile (overlaps load latency)
    stage_lds(cur ^ 1);    // implicit s_wait_loadcnt happens here
    __syncthreads();       // single barrier per K-step
    cur ^= 1;
  }
  compute(cur);            // drain

  // ---- epilogue: C/D layout is M = v + 8*kh, N = r ----------------------
#pragma unroll
  for (int ni = 0; ni < 4; ++ni) {
    const int col = blockN + wc * 64 + ni * 16 + r;
    float bv = 0.0f;
    if constexpr (HAS_BIAS) bv = bias[col];
#pragma unroll
    for (int mi = 0; mi < 2; ++mi) {
      const int row0 = blockM + wr * 32 + mi * 16 + kh * 8;
#pragma unroll
      for (int v = 0; v < 8; ++v) {
        float x = acc[mi][ni][v];
        if constexpr (HAS_BIAS) x += bv;
        if constexpr (RELU) x = fmaxf(x, 0.0f);
        if constexpr (OUT_F32) {
          Cf[(size_t)(row0 + v) * N + col] = x;
        } else {
          Cb[(size_t)(row0 + v) * N + col] = f32_to_bf16_rne(x);
        }
      }
    }
  }
}

// ---------------------------------------------------------------------------
// Host launch
// ---------------------------------------------------------------------------
extern "C" void kernel_launch(void* const* d_in, const int* in_sizes, int n_in,
                              void* d_out, int out_size, void* d_ws, size_t ws_size,
                              hipStream_t stream) {
  const float* x   = (const float*)d_in[0];
  const float* adj = (const float*)d_in[1];
  const float* W1  = (const float*)d_in[2];
  const float* b1  = (const float*)d_in[3];
  const float* W2  = (const float*)d_in[4];
  const float* b2  = (const float*)d_in[5];
  float* out = (float*)d_out;

  constexpr int Nn = 16384, Fin = 512, Hid = 1024, Oc = 256;

  // workspace carve-out (256B aligned)
  char* ws = (char*)d_ws;
  size_t off = 0;
  auto carve = [&](size_t bytes) {
    char* p = ws + off;
    off += (bytes + 255) & ~(size_t)255;
    return p;
  };
  unsigned short* adjb = (unsigned short*)carve((size_t)Nn * Nn * 2);
  unsigned short* xb   = (unsigned short*)carve((size_t)Nn * Fin * 2);
  unsigned short* w1b  = (unsigned short*)carve((size_t)Fin * Hid * 2);
  unsigned short* w2b  = (unsigned short*)carve((size_t)Hid * Oc * 2);
  unsigned short* s1   = (unsigned short*)carve((size_t)Nn * Hid * 2);
  unsigned short* h    = (unsigned short*)carve((size_t)Nn * Hid * 2);
  unsigned short* s2   = (unsigned short*)carve((size_t)Nn * Oc * 2);

  auto cvt = [&](const float* src, unsigned short* dst, long long n) {
    long long n4 = n / 4;
    int blocks = (int)((n4 + 255) / 256);
    cvt_f32_bf16<<<blocks, 256, 0, stream>>>(src, dst, n4);
  };
  cvt(adj, adjb, (long long)Nn * Nn);
  cvt(x,   xb,   (long long)Nn * Fin);
  cvt(W1,  w1b,  (long long)Fin * Hid);
  cvt(W2,  w2b,  (long long)Hid * Oc);

  // s1 = x @ W1
  gcn_gemm_bf16<Hid, Fin, false, false, false>
      <<<dim3(Hid / 128, Nn / 128), 256, 0, stream>>>(
          xb, w1b, nullptr, s1, nullptr);

  // h = relu(adj @ s1 + b1)
  gcn_gemm_bf16<Hid, Nn, true, true, false>
      <<<dim3(Hid / 128, Nn / 128), 256, 0, stream>>>(
          adjb, s1, b1, h, nullptr);

  // s2 = h @ W2
  gcn_gemm_bf16<Oc, Hid, false, false, false>
      <<<dim3(Oc / 128, Nn / 128), 256, 0, stream>>>(
          h, w2b, nullptr, s2, nullptr);

  // out = adj @ s2 + b2   (f32 output)
  gcn_gemm_bf16<Oc, Nn, true, false, true>
      <<<dim3(Oc / 128, Nn / 128), 256, 0, stream>>>(
          adjb, s2, b2, nullptr, out);
}
